// Attention_84327387890534
// MI455X (gfx1250) — matrix operand order repaired and checked by reading, not compiled, not run on hardware
//
#include <hip/hip_runtime.h>
#include <hip/hip_bf16.h>

// ---------------------------------------------------------------------------
// MI455X (gfx1250) fused attention block, bf16 WMMA pipeline.
// ---------------------------------------------------------------------------

typedef __bf16 bf16;
typedef __attribute__((ext_vector_type(16))) bf16  v16bf;
typedef __attribute__((ext_vector_type(8)))  bf16  v8bf;
typedef __attribute__((ext_vector_type(8)))  float v8f;
typedef __attribute__((ext_vector_type(4)))  int   v4i;

#define DIMC   1024
#define HEADS  16
#define HD     64
#define NBATCH 8
#define NTOK   1024
#define QK_SCALE 0.125f   // 64^-0.5

// ---- async global->LDS (verified present on this toolchain) ---------------
#if defined(__has_builtin)
#  if __has_builtin(__builtin_amdgcn_global_load_async_to_lds_b128)
#    define HAVE_ASYNC_LDS 1
#  endif
#  if __has_builtin(__builtin_amdgcn_s_wait_asynccnt)
#    define HAVE_WAIT_ASYNC 1
#  endif
#endif

__device__ __forceinline__ void wait_async0() {
#ifdef HAVE_WAIT_ASYNC
  __builtin_amdgcn_s_wait_asynccnt(0);
#else
  asm volatile("s_wait_asynccnt 0x0" ::: "memory");
#endif
}

#ifdef HAVE_ASYNC_LDS
__device__ __forceinline__ void cp16_async(bf16* dst, const bf16* src) {
  __builtin_amdgcn_global_load_async_to_lds_b128(
      (__attribute__((address_space(1))) v4i*)(const void*)src,
      (__attribute__((address_space(3))) v4i*)(void*)dst, 0, 0);
}
#endif

__device__ __forceinline__ v8f wmma_bf16(v16bf a, v16bf b, v8f c) {
  return __builtin_amdgcn_wmma_f32_16x16x32_bf16(false, a, false, b, (short)0, c,
                                                 false, false);
}

// A-matrix 16x32 bf16 fragment: lane<16 holds K={0..7,16..23}, lane>=16 the rest.
__device__ __forceinline__ v16bf load_afrag(const bf16* row, int khalf) {
  v8bf lo = *(const v8bf*)(row + khalf * 8);
  v8bf hi = *(const v8bf*)(row + 16 + khalf * 8);
  v16bf r;
#pragma unroll
  for (int j = 0; j < 8; ++j) { r[j] = lo[j]; r[8 + j] = hi[j]; }
  return r;
}

// B-matrix 32x16 bf16 fragment: lane<16 holds K={0..7,16..23}, lane>=16 the rest.
__device__ __forceinline__ v16bf load_bfrag(const bf16* col, int khalf) {
  v8bf lo = *(const v8bf*)(col + khalf * 8);
  v8bf hi = *(const v8bf*)(col + 16 + khalf * 8);
  v16bf r;
#pragma unroll
  for (int j = 0; j < 8; ++j) { r[j] = lo[j]; r[8 + j] = hi[j]; }
  return r;
}

// ---------------------------------------------------------------------------
// Prep kernels
// ---------------------------------------------------------------------------
__global__ void k_prep_x(const float* __restrict__ x, const float* __restrict__ noise,
                         const float* __restrict__ ns, bf16* __restrict__ xb, int n) {
  int i = blockIdx.x * blockDim.x + threadIdx.x;
  if (i < n) xb[i] = (bf16)(x[i] + noise[i >> 10] * ns[0]);
}

__global__ void k_transpose_w(const float* __restrict__ w, bf16* __restrict__ wt,
                              int rows, int cols) {
  int i = blockIdx.x * blockDim.x + threadIdx.x;
  if (i < rows * cols) {
    int c = i / rows, k = i - c * rows;
    wt[i] = (bf16)w[k * cols + c];
  }
}

// ---------------------------------------------------------------------------
// QKV GEMM: 32x64 tile per wave (2 A-frags x 4 B-frags, 8 wmma / K-step).
// ---------------------------------------------------------------------------
__global__ void __launch_bounds__(256) k_qkv(const bf16* __restrict__ xb,
                                             const bf16* __restrict__ wqt,
                                             bf16* __restrict__ qb,
                                             bf16* __restrict__ kb,
                                             bf16* __restrict__ vtb) {
  const int NTN = 3 * DIMC / 64;  // 48 col tiles
  int wave = (blockIdx.x * blockDim.x + threadIdx.x) >> 5;
  int lane = threadIdx.x & 31;
  int mt = wave / NTN, nt = wave - mt * NTN;   // mt: 32-row tile, nt: 64-col tile
  int n0 = lane & 15, half = lane >> 4;

  const bf16* arow0 = xb + (size_t)(mt * 32 + n0) * DIMC;
  const bf16* arow1 = arow0 + 16 * DIMC;
  const bf16* brow0 = wqt + (size_t)(nt * 64 + n0) * DIMC;

  v8f acc[2][4] = {};
#pragma unroll 2
  for (int k = 0; k < DIMC; k += 32) {
    // locality 3 -> near (WGP) scope prefetch; pulls next lines into L0/WGP$
    __builtin_prefetch(arow0 + k + 512, 0, 3);
    __builtin_prefetch(brow0 + k + 512, 0, 3);
    v16bf a0 = load_afrag(arow0 + k, half);
    v16bf a1 = load_afrag(arow1 + k, half);
#pragma unroll
    for (int c = 0; c < 4; ++c) {
      v16bf bf = load_bfrag(brow0 + (size_t)c * 16 * DIMC + k, half);
      acc[0][c] = wmma_bf16(a0, bf, acc[0][c]);
      acc[1][c] = wmma_bf16(a1, bf, acc[1][c]);
    }
  }

  int sec = nt >> 4;   // 0=q 1=k 2=v  (uniform per wave)
  int h   = nt & 15;   // head          (uniform per wave)
  if (sec == 2) {
    // v transposed: per lane d = c*16+n0 fixed, 8 consecutive tokens -> b128
#pragma unroll
    for (int i = 0; i < 2; ++i)
#pragma unroll
      for (int c = 0; c < 4; ++c) {
        int m0 = mt * 32 + i * 16 + half * 8;
        int b = m0 >> 10, t0 = m0 & 1023;
        v8bf pk;
#pragma unroll
        for (int r = 0; r < 8; ++r) pk[r] = (bf16)acc[i][c][r];
        *(v8bf*)(vtb + ((size_t)(b * HEADS + h) * HD + c * 16 + n0) * NTOK + t0) = pk;
      }
  } else {
    bf16* dst = (sec == 0) ? qb : kb;
    float scl = (sec == 0) ? QK_SCALE : 1.0f;
#pragma unroll
    for (int i = 0; i < 2; ++i)
#pragma unroll
      for (int c = 0; c < 4; ++c)
#pragma unroll
        for (int r = 0; r < 8; ++r) {
          int m = mt * 32 + i * 16 + half * 8 + r;
          int b = m >> 10, t = m & 1023;
          dst[((size_t)(b * HEADS + h) * NTOK + t) * HD + c * 16 + n0] =
              (bf16)(acc[i][c][r] * scl);
        }
  }
}

// ---------------------------------------------------------------------------
// Flash attention: 4 waves/block share one (b,h); K/V tiles staged in LDS
// via async global->LDS, one 16-row q tile per wave.
// ---------------------------------------------------------------------------
__global__ void __launch_bounds__(128) k_flash(const bf16* __restrict__ qb,
                                               const bf16* __restrict__ kb,
                                               const bf16* __restrict__ vtb,
                                               const float* __restrict__ bias_table,
                                               bf16* __restrict__ ob) {
  __shared__ bf16 kbuf[32 * 64];      // 32 kv tokens x 64 hd
  __shared__ bf16 vbuf[64 * 32];      // 64 hd rows  x 32 kv tokens (v^T)
  __shared__ bf16 pbuf[4][16 * 32];   // per-wave P re-layout

  int tid = threadIdx.x;
  int wid = tid >> 5, lane = tid & 31;
  int qt = blockIdx.x * 4 + wid;      // 4 | 64 => same bh for whole block
  int bh = qt >> 6, rt = qt & 63;
  int h = bh & 15, b = bh >> 4;
  int n0 = lane & 15, half = lane >> 4;

  const bf16* qp = qb  + (size_t)bh * NTOK * HD;
  const bf16* kp = kb  + (size_t)bh * NTOK * HD;
  const bf16* vp = vtb + (size_t)bh * HD * NTOK;
  bf16* pw = pbuf[wid];

  const bf16* aq = qp + (size_t)(rt * 16 + n0) * HD;
  v16bf qf0 = load_afrag(aq + 0,  half);
  v16bf qf1 = load_afrag(aq + 32, half);

  v8f acc0 = {}, acc1 = {}, acc2 = {}, acc3 = {};
  float mrow[8], lrow[8];
#pragma unroll
  for (int r = 0; r < 8; ++r) { mrow[r] = -1e30f; lrow[r] = 0.f; }

  // block-cooperative staging indices (128 threads)
  int krow = tid >> 2, kch = (tid & 3) << 4;   // k: 32 rows x 64
  int vrow = tid >> 1, vch = (tid & 1) << 4;   // v: 64 rows x 32

  for (int kv0 = 0; kv0 < NTOK; kv0 += 32) {
    const bf16* ksrc = kp + (size_t)(kv0 + krow) * HD + kch;
    const bf16* vsrc = vp + (size_t)vrow * NTOK + kv0 + vch;
    bf16* kdst = kbuf + krow * 64 + kch;
    bf16* vdst = vbuf + vrow * 32 + vch;
#ifdef HAVE_ASYNC_LDS
    cp16_async(kdst,     ksrc);
    cp16_async(kdst + 8, ksrc + 8);
    cp16_async(vdst,     vsrc);
    cp16_async(vdst + 8, vsrc + 8);
    wait_async0();
#else
    *(v8bf*)kdst       = *(const v8bf*)ksrc;
    *(v8bf*)(kdst + 8) = *(const v8bf*)(ksrc + 8);
    *(v8bf*)vdst       = *(const v8bf*)vsrc;
    *(v8bf*)(vdst + 8) = *(const v8bf*)(vsrc + 8);
#endif
    __syncthreads();

    // S = q * k^T for a 16x32 chunk (K tiles from LDS)
    v8f s0 = {}, s1 = {};
    {
      const bf16* bk0 = kbuf + n0 * 64;
      const bf16* bk1 = kbuf + (16 + n0) * 64;
      s0 = wmma_bf16(qf0, load_bfrag(bk0 + 0,  half), s0);
      s0 = wmma_bf16(qf1, load_bfrag(bk0 + 32, half), s0);
      s1 = wmma_bf16(qf0, load_bfrag(bk1 + 0,  half), s1);
      s1 = wmma_bf16(qf1, load_bfrag(bk1 + 32, half), s1);
    }

    // online softmax; rows m = 8*half+r live across 16 lanes, col = n0
#pragma unroll
    for (int r = 0; r < 8; ++r) {
      int i = rt * 16 + 8 * half + r;
      float v0 = s0[r] + bias_table[(i - (kv0 + n0)      + 1023) * HEADS + h];
      float v1 = s1[r] + bias_table[(i - (kv0 + 16 + n0) + 1023) * HEADS + h];
      float mx = fmaxf(v0, v1);
#pragma unroll
      for (int msk = 1; msk < 16; msk <<= 1) mx = fmaxf(mx, __shfl_xor(mx, msk, 32));
      float mnew = fmaxf(mrow[r], mx);
      float corr = __expf(mrow[r] - mnew);
      float p0 = __expf(v0 - mnew);
      float p1 = __expf(v1 - mnew);
      float ps = p0 + p1;
#pragma unroll
      for (int msk = 1; msk < 16; msk <<= 1) ps += __shfl_xor(ps, msk, 32);
      lrow[r] = lrow[r] * corr + ps;
      mrow[r] = mnew;
      acc0[r] *= corr; acc1[r] *= corr; acc2[r] *= corr; acc3[r] *= corr;
      pw[(8 * half + r) * 32 + n0]      = (bf16)p0;
      pw[(8 * half + r) * 32 + 16 + n0] = (bf16)p1;
    }

    // P (via LDS re-layout) x V (from LDS)
    v16bf pf = load_afrag(pw + n0 * 32, half);
    acc0 = wmma_bf16(pf, load_bfrag(vbuf + ( 0 + n0) * 32, half), acc0);
    acc1 = wmma_bf16(pf, load_bfrag(vbuf + (16 + n0) * 32, half), acc1);
    acc2 = wmma_bf16(pf, load_bfrag(vbuf + (32 + n0) * 32, half), acc2);
    acc3 = wmma_bf16(pf, load_bfrag(vbuf + (48 + n0) * 32, half), acc3);
    __syncthreads();
  }

#pragma unroll
  for (int r = 0; r < 8; ++r) {
    int i = rt * 16 + 8 * half + r;
    float inv = 1.0f / lrow[r];
    size_t base = ((size_t)(b * NTOK + i)) * DIMC + h * HD;
    ob[base + 0  + n0] = (bf16)(acc0[r] * inv);
    ob[base + 16 + n0] = (bf16)(acc1[r] * inv);
    ob[base + 32 + n0] = (bf16)(acc2[r] * inv);
    ob[base + 48 + n0] = (bf16)(acc3[r] * inv);
  }
}

// ---------------------------------------------------------------------------
// Output projection: 32x64 tile per wave + bias, fp32 out
// ---------------------------------------------------------------------------
__global__ void __launch_bounds__(256) k_proj(const bf16* __restrict__ ob,
                                              const bf16* __restrict__ wpt,
                                              const float* __restrict__ bproj,
                                              float* __restrict__ out) {
  const int NTN = DIMC / 64;  // 16
  int wave = (blockIdx.x * blockDim.x + threadIdx.x) >> 5;
  int lane = threadIdx.x & 31;
  int mt = wave / NTN, nt = wave - mt * NTN;
  int n0 = lane & 15, half = lane >> 4;

  const bf16* arow0 = ob + (size_t)(mt * 32 + n0) * DIMC;
  const bf16* arow1 = arow0 + 16 * DIMC;
  const bf16* brow0 = wpt + (size_t)(nt * 64 + n0) * DIMC;

  v8f acc[2][4] = {};
#pragma unroll 2
  for (int k = 0; k < DIMC; k += 32) {
    __builtin_prefetch(arow0 + k + 512, 0, 3);
    __builtin_prefetch(brow0 + k + 512, 0, 3);
    v16bf a0 = load_afrag(arow0 + k, half);
    v16bf a1 = load_afrag(arow1 + k, half);
#pragma unroll
    for (int c = 0; c < 4; ++c) {
      v16bf bf = load_bfrag(brow0 + (size_t)c * 16 * DIMC + k, half);
      acc[0][c] = wmma_bf16(a0, bf, acc[0][c]);
      acc[1][c] = wmma_bf16(a1, bf, acc[1][c]);
    }
  }

#pragma unroll
  for (int i = 0; i < 2; ++i)
#pragma unroll
    for (int c = 0; c < 4; ++c) {
      int n = nt * 64 + c * 16 + n0;
      float bias = bproj[n];
#pragma unroll
      for (int r = 0; r < 8; ++r)
        out[(size_t)(mt * 32 + i * 16 + 8 * half + r) * DIMC + n] = acc[i][c][r] + bias;
    }
}

// ---------------------------------------------------------------------------
// Launch
// ---------------------------------------------------------------------------
extern "C" void kernel_launch(void* const* d_in, const int* in_sizes, int n_in,
                              void* d_out, int out_size, void* d_ws, size_t ws_size,
                              hipStream_t stream) {
  const float* x     = (const float*)d_in[0];
  const float* noise = (const float*)d_in[1];
  const float* w_qkv = (const float*)d_in[2];
  const float* w_prj = (const float*)d_in[3];
  const float* b_prj = (const float*)d_in[4];
  const float* bias  = (const float*)d_in[5];
  const float* ns    = (const float*)d_in[6];
  float* out = (float*)d_out;

  char* ws = (char*)d_ws;
  const size_t SZ_X   = (size_t)NBATCH * NTOK * DIMC * 2;       // 16 MB
  const size_t SZ_WQ  = (size_t)3 * DIMC * DIMC * 2;            //  6 MB
  const size_t SZ_WP  = (size_t)DIMC * DIMC * 2;                //  2 MB
  const size_t SZ_QKV = (size_t)NBATCH * HEADS * NTOK * HD * 2; // 16 MB each
  bf16* xb  = (bf16*)(ws);
  bf16* wqt = (bf16*)(ws + SZ_X);
  bf16* wpt = (bf16*)(ws + SZ_X + SZ_WQ);
  bf16* qb  = (bf16*)(ws + SZ_X + SZ_WQ + SZ_WP);
  bf16* kb  = (bf16*)(ws + SZ_X + SZ_WQ + SZ_WP + SZ_QKV);
  bf16* vtb = (bf16*)(ws + SZ_X + SZ_WQ + SZ_WP + 2 * SZ_QKV);
  bf16* ob  = xb;  // xb dead after k_qkv

  const int NX = NBATCH * NTOK * DIMC;
  k_prep_x<<<NX / 256, 256, 0, stream>>>(x, noise, ns, xb, NX);
  k_transpose_w<<<(3 * DIMC * DIMC) / 256, 256, 0, stream>>>(w_qkv, wqt, DIMC, 3 * DIMC);
  k_transpose_w<<<(DIMC * DIMC) / 256, 256, 0, stream>>>(w_prj, wpt, DIMC, DIMC);

  // 256 row-tiles x 48 col-tiles, 8 waves/block
  k_qkv<<<(256 * 48) / 8, 256, 0, stream>>>(xb, wqt, qb, kb, vtb);

  // 8192 q-tiles, 4 waves/block (block shares one (b,h))
  k_flash<<<8192 / 4, 128, 0, stream>>>(qb, kb, vtb, bias, ob);

  // 256 x 16 tiles, 8 waves/block
  k_proj<<<(256 * 16) / 8, 256, 0, stream>>>(ob, wpt, b_prj, out);
}
